// GNN_76785425318278
// MI455X (gfx1250) — matrix operand order repaired and checked by reading, not compiled, not run on hardware
//
#include <hip/hip_runtime.h>
#include <hip/hip_bf16.h>

#define N_NODES 50000
#define N_EDGES 800000
#define D 128
#define N_LAYERS 9
#define N_GRAPHS 256
#define EPS 1e-5f

typedef __bf16 v16bf __attribute__((ext_vector_type(16)));
typedef float  v8f   __attribute__((ext_vector_type(8)));

__device__ __forceinline__ __bf16 f2bf(float f) {
    unsigned u = __builtin_bit_cast(unsigned, f);
    u += 0x7FFFu + ((u >> 16) & 1u);               // round-to-nearest-even
    unsigned short s = (unsigned short)(u >> 16);
    return __builtin_bit_cast(__bf16, s);
}

// Hardware f32 scatter-add (no CAS fallback): GLOBAL_ATOMIC_ADD_F32, no return.
__device__ __forceinline__ void gatomic_add_f32(float* p, float v) {
    asm volatile("global_atomic_add_f32 %0, %1, off" :: "v"(p), "v"(v) : "memory");
}

// CDNA5 async DMA: 16B global -> LDS without touching VGPRs (tracked by ASYNCcnt).
__device__ __forceinline__ void async_ld_b128(unsigned lds_byte_addr, const void* gaddr) {
    asm volatile("global_load_async_to_lds_b128 %0, %1, off"
                 :: "v"(lds_byte_addr), "v"(gaddr) : "memory");
}
__device__ __forceinline__ void wait_async0() {
    asm volatile("s_wait_asynccnt 0x0" ::: "memory");
}
__device__ __forceinline__ unsigned lds_addr_of(const void* p) {
    return (unsigned)(unsigned long long)(uintptr_t)p;   // low 32 bits = LDS offset
}

__global__ void k_fill4(float4* __restrict__ p, int n4) {
    int i = blockIdx.x * 256 + threadIdx.x;
    if (i < n4) p[i] = make_float4(0.f, 0.f, 0.f, 0.f);
}

// conv_w (9 x 128 x 128, f32) -> bf16 once per launch.
__global__ void k_wprep(const float* __restrict__ w, __bf16* __restrict__ wbf, int n) {
    int i = blockIdx.x * 256 + threadIdx.x;
    if (i < n) wbf[i] = f2bf(w[i]);
}

// h = x @ Wv   [50000,13] x [13,128]
__global__ void __launch_bounds__(256) k_init_h(const float* __restrict__ x,
                                                const float* __restrict__ Wv,
                                                float* __restrict__ h) {
    int i = blockIdx.x * 256 + threadIdx.x;      // 6.4M elements, exact grid
    int n = i >> 7, j = i & 127;
    float s = 0.f;
#pragma unroll
    for (int k = 0; k < 13; ++k) s += x[n * 13 + k] * Wv[k * 128 + j];
    h[i] = s;
}

// Edge stage: msg = relu(h[src] + edge_attr@We) scatter-added into aggr[dst].
// One wave32 handles 4 consecutive edges; each lane owns 4 channels (float4).
__global__ void __launch_bounds__(256) k_edges(const float* __restrict__ h,
                                               const int* __restrict__ ei,
                                               const float* __restrict__ eattr,
                                               const float* __restrict__ We,
                                               float* __restrict__ aggr) {
    __shared__ float sWe[4 * 128];
    int tid = threadIdx.x;
    sWe[tid]       = We[tid];
    sWe[tid + 256] = We[tid + 256];
    __syncthreads();

    int wave = tid >> 5, lane = tid & 31;
    int ch = lane * 4;
    int e0 = blockIdx.x * 32 + wave * 4;         // 800000 % 32 == 0, no guard
    if (e0 + 512 < N_EDGES) __builtin_prefetch(&eattr[(size_t)(e0 + 512) * 4], 0, 0);

#pragma unroll
    for (int q = 0; q < 4; ++q) {
        int e = e0 + q;
        int src = ei[e];
        int dst = ei[N_EDGES + e];
        float a0 = eattr[e * 4 + 0], a1 = eattr[e * 4 + 1];
        float a2 = eattr[e * 4 + 2], a3 = eattr[e * 4 + 3];

        const float4 hv = *(const float4*)(h + (size_t)src * D + ch);
        float m0 = hv.x + a0 * sWe[ch + 0] + a1 * sWe[128 + ch + 0] + a2 * sWe[256 + ch + 0] + a3 * sWe[384 + ch + 0];
        float m1 = hv.y + a0 * sWe[ch + 1] + a1 * sWe[128 + ch + 1] + a2 * sWe[256 + ch + 1] + a3 * sWe[384 + ch + 1];
        float m2 = hv.z + a0 * sWe[ch + 2] + a1 * sWe[128 + ch + 2] + a2 * sWe[256 + ch + 2] + a3 * sWe[384 + ch + 2];
        float m3 = hv.w + a0 * sWe[ch + 3] + a1 * sWe[128 + ch + 3] + a2 * sWe[256 + ch + 3] + a3 * sWe[384 + ch + 3];
        m0 = fmaxf(m0, 0.f); m1 = fmaxf(m1, 0.f); m2 = fmaxf(m2, 0.f); m3 = fmaxf(m3, 0.f);

        float* ap = aggr + (size_t)dst * D + ch;
        gatomic_add_f32(ap + 0, m0);
        gatomic_add_f32(ap + 1, m1);
        gatomic_add_f32(ap + 2, m2);
        gatomic_add_f32(ap + 3, m3);
    }
}

// Dense layer: h = relu(((1+eps)h + aggr) @ W + b) + h, in place (block-local rows).
// W (pre-converted bf16) is DMA'd into LDS with global_load_async_to_lds_b128,
// overlapped with the A-tile build; then 8 waves run bf16 WMMA 16x16x32, K=128.
__global__ void __launch_bounds__(256) k_gemm(float* __restrict__ h,
                                              const float* __restrict__ aggr,
                                              const __bf16* __restrict__ Wg,
                                              const float* __restrict__ bias) {
    __shared__ __align__(16) __bf16 sW[128 * 128];   // W as [k][n], bf16 (32 KB)
    __shared__ __align__(16) __bf16 sA[16 * 128];    // A tile as [m][k], bf16 (4 KB)
    int tid = threadIdx.x;
    int rowBase = blockIdx.x * 16;     // 50000/16 = 3125 blocks exactly

    // Kick off async W copy: 32 KB = 2048 x b128, 8 per thread.
    unsigned ldsW = lds_addr_of(&sW[0]);
#pragma unroll
    for (int i = 0; i < 8; ++i) {
        int byteOff = (tid + 256 * i) * 16;
        async_ld_b128(ldsW + byteOff, (const char*)Wg + byteOff);
    }

    // Overlap: build bf16 A tile = (1+eps)*h + aggr.
#pragma unroll
    for (int i = 0; i < 8; ++i) {
        int idx = tid + 256 * i;                       // [m][k] flat, rows contiguous
        size_t gi = (size_t)rowBase * 128 + idx;
        sA[idx] = f2bf((1.0f + EPS) * h[gi] + aggr[gi]);
    }
    wait_async0();
    __syncthreads();

    int lane = tid & 31, wave = tid >> 5;
    int n0  = wave * 16;
    int m   = lane & 15;
    int kA0 = (lane >> 4) * 8;                 // A: lanes 16-31 hold K+8 group
    int kB0 = (lane & 15) + (lane >> 4) * 16;  // unused: B below indexes K via kA0, same order as A

    v8f c = {};
#pragma unroll
    for (int ks = 0; ks < 128; ks += 32) {
        v16bf af, bfr;
#pragma unroll
        for (int i = 0; i < 16; ++i)       // A 16x32 bf16 VGPR layout (ISA 7.12.2)
            af[i] = sA[m * 128 + ks + (i & 7) + ((i >> 3) << 4) + kA0];
#pragma unroll
        for (int i = 0; i < 16; ++i)       // B 32x16: lane->N, slot->K (same K order as A)
            bfr[i] = sW[(ks + kA0 + (i & 7) + ((i >> 3) << 4)) * 128 + n0 + (lane & 15)];
        c = __builtin_amdgcn_wmma_f32_16x16x32_bf16(false, af, false, bfr,
                                                    (short)0, c, false, false);
    }

    // C/D layout: N = lane&15; lanes<16 -> M = r, lanes>=16 -> M = r+8.
    int col = n0 + (lane & 15);
    float b = bias[col];
    int mB = (lane >> 4) * 8;
#pragma unroll
    for (int r = 0; r < 8; ++r) {
        size_t o = (size_t)(rowBase + mB + r) * 128 + col;
        float v = fmaxf(c[r] + b, 0.f);
        h[o] = v + h[o];                  // residual: same lane reads then writes
    }
}

// Graph mean pool (sums + counts via HW atomics). One wave per node.
__global__ void __launch_bounds__(256) k_pool(const float* __restrict__ h,
                                              const int* __restrict__ batch,
                                              float* __restrict__ pooled,
                                              float* __restrict__ cnt) {
    int tid = threadIdx.x;
    int wave = tid >> 5, lane = tid & 31;
    int n = blockIdx.x * 8 + wave;        // 50000/8 = 6250 blocks exactly
    int g = batch[n];
    int ch = lane * 4;
    const float4 hv = *(const float4*)(h + (size_t)n * D + ch);
    float* pp = pooled + (size_t)g * D + ch;
    gatomic_add_f32(pp + 0, hv.x);
    gatomic_add_f32(pp + 1, hv.y);
    gatomic_add_f32(pp + 2, hv.z);
    gatomic_add_f32(pp + 3, hv.w);
    if (lane == 0) gatomic_add_f32(cnt + g, 1.0f);
}

// Head MLP: gelu(pooled@Wh1 + bh1) @ Wh2 + bh2, one block per graph.
__global__ void __launch_bounds__(128) k_mlp(const float* __restrict__ pooled,
                                             const float* __restrict__ cnt,
                                             const float* __restrict__ Wh1,
                                             const float* __restrict__ bh1,
                                             const float* __restrict__ Wh2,
                                             const float* __restrict__ bh2,
                                             float* __restrict__ out) {
    __shared__ float sp[128];
    __shared__ float red[128];
    int t = threadIdx.x, g = blockIdx.x;
    float c = fmaxf(cnt[g], 1.0f);
    sp[t] = pooled[(size_t)g * 128 + t] / c;
    __syncthreads();
    float acc = 0.f;
#pragma unroll
    for (int jj = 0; jj < 4; ++jj) {
        int j = t + jj * 128;             // 512 hidden units
        float d = bh1[j];
        for (int k = 0; k < 128; ++k) d += sp[k] * Wh1[k * 512 + j];
        float ge = 0.5f * d * (1.0f + erff(d * 0.7071067811865475f));   // exact gelu
        acc += ge * Wh2[j];
    }
    red[t] = acc;
    __syncthreads();
    for (int s = 64; s > 0; s >>= 1) {
        if (t < s) red[t] += red[t + s];
        __syncthreads();
    }
    if (t == 0) out[g] = red[0] + bh2[0];
}

extern "C" void kernel_launch(void* const* d_in, const int* in_sizes, int n_in,
                              void* d_out, int out_size, void* d_ws, size_t ws_size,
                              hipStream_t stream) {
    const float* x     = (const float*)d_in[0];
    const int*   ei    = (const int*)  d_in[1];
    const float* eattr = (const float*)d_in[2];
    const int*   batch = (const int*)  d_in[3];
    const float* Wv    = (const float*)d_in[4];
    const float* We    = (const float*)d_in[5];
    const float* convW = (const float*)d_in[6];
    const float* convB = (const float*)d_in[7];
    const float* Wh1   = (const float*)d_in[8];
    const float* bh1   = (const float*)d_in[9];
    const float* Wh2   = (const float*)d_in[10];
    const float* bh2   = (const float*)d_in[11];
    float* out = (float*)d_out;

    float*  h      = (float*)d_ws;                     // 50000*128 f32
    float*  aggr   = h + (size_t)N_NODES * D;          // 50000*128 f32
    float*  pooled = aggr + (size_t)N_NODES * D;       // 256*128 f32
    float*  cnt    = pooled + (size_t)N_GRAPHS * D;    // 256 f32 (contiguous)
    __bf16* Wbf    = (__bf16*)(cnt + N_GRAPHS);        // 9*128*128 bf16 (16B-aligned)

    const int NWH = N_LAYERS * D * D;                  // 147456
    k_wprep<<<(NWH + 255) / 256, 256, 0, stream>>>(convW, Wbf, NWH);
    k_init_h<<<(N_NODES * D) / 256, 256, 0, stream>>>(x, Wv, h);

    for (int L = 0; L < N_LAYERS; ++L) {
        k_fill4<<<(N_NODES * D / 4) / 256, 256, 0, stream>>>((float4*)aggr, N_NODES * D / 4);
        k_edges<<<N_EDGES / 32, 256, 0, stream>>>(h, ei, eattr, We, aggr);
        k_gemm<<<N_NODES / 16, 256, 0, stream>>>(h, aggr,
                                                 Wbf + (size_t)L * D * D,
                                                 convB + (size_t)L * D);
    }

    int tail4 = (N_GRAPHS * D + N_GRAPHS) / 4;         // pooled + cnt, contiguous
    k_fill4<<<(tail4 + 255) / 256, 256, 0, stream>>>((float4*)pooled, tail4);
    k_pool<<<N_NODES / 8, 256, 0, stream>>>(h, batch, pooled, cnt);
    k_mlp<<<N_GRAPHS, 128, 0, stream>>>(pooled, cnt, Wh1, bh1, Wh2, bh2, out);
}
